// SelfAttention_15994458210628
// MI455X (gfx1250) — matrix operand order repaired and checked by reading, not compiled, not run on hardware
//
#include <hip/hip_runtime.h>

// ---------------- problem constants ----------------
constexpr int Bn  = 2;
constexpr int Sn  = 2048;
constexpr int DMn = 1024;
constexpr int Hn  = 16;
constexpr int Dn  = 64;        // head dim
// SCALE = 1/sqrt(64) = 0.125

typedef __attribute__((ext_vector_type(16))) __bf16        v16bf;
typedef __attribute__((ext_vector_type(8)))  float         v8f;
typedef __attribute__((ext_vector_type(8)))  unsigned int  v8u;

#define WMMA_BF16(a, b, c) \
  __builtin_amdgcn_wmma_f32_16x16x32_bf16(false, (a), false, (b), (short)0, (c), false, false)

// fp32 -> bf16, round-to-nearest-even
__device__ __forceinline__ unsigned short f2bf(float f) {
  unsigned int u = __builtin_bit_cast(unsigned int, f);
  return (unsigned short)((u + 0x7fffu + ((u >> 16) & 1u)) >> 16);
}

// A-matrix 16x32 bf16 fragment (ISA 7.12.2): lane holds row m=lane&15,
// VGPR v holds k-pair at 16*(v>>2) + 8*half + 2*(v&3).
__device__ __forceinline__ v16bf load_fragA(const unsigned short* p, int ld, int lane) {
  const int lh = lane & 15, hf = (lane >> 4) & 1;
  const unsigned short* rp = p + lh * ld;
  v8u u;
#pragma unroll
  for (int v = 0; v < 8; ++v) {
    const int k = ((v >> 2) << 4) + (hf << 3) + ((v & 3) << 1);
    u[v] = *reinterpret_cast<const unsigned int*>(rp + k);
  }
  return __builtin_bit_cast(v16bf, u);
}

// B-matrix 32x16 bf16 fragment: lane holds column n=lane&15,
// VGPR v holds k-pair at 16*(v>>2) + 8*half + 2*(v&3) (same k order as A).
// p points at element (n0, k0) of a row-major [N][K] operand (i.e. B^T storage).
__device__ __forceinline__ v16bf load_fragB(const unsigned short* p, int ld, int lane) {
  const int lh = lane & 15, hf = (lane >> 4) & 1;
  const unsigned short* rp = p + lh * ld;
  v8u u;
#pragma unroll
  for (int v = 0; v < 8; ++v) {
    const int k = ((v >> 2) << 4) + (hf << 3) + ((v & 3) << 1);
    u[v] = *reinterpret_cast<const unsigned int*>(rp + k);
  }
  return __builtin_bit_cast(v16bf, u);
}

// ---------------- conversion kernels ----------------
__global__ void cvt_bf16_kernel(const float* __restrict__ in, unsigned short* __restrict__ out, int n) {
  const int i = blockIdx.x * blockDim.x + threadIdx.x;
  if (i < n) out[i] = f2bf(in[i]);
}

// in: [Kd][Nd] f32 row-major  ->  out: [Nd][Kd] bf16 (transposed)
__global__ void cvt_T_bf16_kernel(const float* __restrict__ in, unsigned short* __restrict__ out,
                                  int Kd, int Nd) {
  const int i = blockIdx.x * blockDim.x + threadIdx.x;
  if (i < Kd * Nd) {
    const int n = i / Kd, k = i % Kd;
    out[i] = f2bf(in[k * Nd + n]);
  }
}

// ---------------- QKV projection GEMM ----------------
// One wave computes a 16(M) x 64(N) strip. Ping-pong double buffering (2 k-steps
// per iteration, no cross-buffer copies): WMMAs overlap the other buffer's loads.
__global__ void __launch_bounds__(128) qkv_gemm_kernel(
    const unsigned short* __restrict__ X,    // [B*S][DM] bf16
    const unsigned short* __restrict__ Wt,   // [3*DM][DM] bf16 (w_qkv transposed)
    const float* __restrict__ bias,          // [3*DM]
    unsigned short* __restrict__ Q,
    unsigned short* __restrict__ K,
    unsigned short* __restrict__ Vt) {
  const int lane = threadIdx.x & 31;
  const int wid  = blockIdx.x * 4 + (threadIdx.x >> 5);
  const int mt = wid / 48, ns = wid % 48;          // 256 x 48 strips
  const int m0 = mt << 4, n0 = ns << 6;
  const int lh = lane & 15, hf = lane >> 4;
  const unsigned short* Xp = X + (size_t)m0 * DMn;

  v8f acc[4] = {};
  v16bf a0 = load_fragA(Xp, DMn, lane);
  v16bf b0[4];
#pragma unroll
  for (int nt = 0; nt < 4; ++nt)
    b0[nt] = load_fragB(Wt + (size_t)(n0 + nt * 16) * DMn, DMn, lane);

  for (int kk = 0; kk < DMn; kk += 64) {
    // fill buffer 1 (k-step kk+32) while computing on buffer 0
    const v16bf a1 = load_fragA(Xp + kk + 32, DMn, lane);
    v16bf b1[4];
#pragma unroll
    for (int nt = 0; nt < 4; ++nt)
      b1[nt] = load_fragB(Wt + (size_t)(n0 + nt * 16) * DMn + kk + 32, DMn, lane);
#pragma unroll
    for (int nt = 0; nt < 4; ++nt)
      acc[nt] = WMMA_BF16(a0, b0[nt], acc[nt]);
    // refill buffer 0 (k-step kk+64) while computing on buffer 1
    if (kk + 64 < DMn) {
      a0 = load_fragA(Xp + kk + 64, DMn, lane);
#pragma unroll
      for (int nt = 0; nt < 4; ++nt)
        b0[nt] = load_fragB(Wt + (size_t)(n0 + nt * 16) * DMn + kk + 64, DMn, lane);
    }
#pragma unroll
    for (int nt = 0; nt < 4; ++nt)
      acc[nt] = WMMA_BF16(a1, b1[nt], acc[nt]);
  }

#pragma unroll
  for (int nt = 0; nt < 4; ++nt) {
    const float bv = bias[n0 + nt * 16 + lh];
#pragma unroll
    for (int v = 0; v < 8; ++v) acc[nt][v] += bv;
  }

  const int bidx  = m0 >> 11;            // batch (S=2048 rows per batch)
  const int sbase = m0 & (Sn - 1);
  const int h     = (n0 & (DMn - 1)) >> 6;
  if (n0 < DMn) {                        // ---- Q (pre-scaled) ----
    unsigned short* dst = Q + (size_t)(bidx * Hn + h) * Sn * Dn;
#pragma unroll
    for (int nt = 0; nt < 4; ++nt)
#pragma unroll
      for (int v = 0; v < 8; ++v)
        dst[(size_t)(sbase + v + (hf << 3)) * Dn + nt * 16 + lh] = f2bf(acc[nt][v] * 0.125f);
  } else if (n0 < 2 * DMn) {             // ---- K ----
    unsigned short* dst = K + (size_t)(bidx * Hn + h) * Sn * Dn;
#pragma unroll
    for (int nt = 0; nt < 4; ++nt)
#pragma unroll
      for (int v = 0; v < 8; ++v)
        dst[(size_t)(sbase + v + (hf << 3)) * Dn + nt * 16 + lh] = f2bf(acc[nt][v]);
  } else {                               // ---- V transposed ----
    unsigned short* dst = Vt + (size_t)(bidx * Hn + h) * Dn * Sn;
#pragma unroll
    for (int nt = 0; nt < 4; ++nt)
#pragma unroll
      for (int v = 0; v < 8; ++v)
        dst[(size_t)(nt * 16 + lh) * Sn + sbase + v + (hf << 3)] = f2bf(acc[nt][v]);
  }
}

// ---------------- causal flash attention ----------------
// One wave per (b,h, 32-row q-block): 2 row-tiles x 64 d. Online softmax over
// 32-key blocks (q-tiles align with key blocks -> exact causal trip count).
// V-fragment loads are hoisted ahead of the softmax so its VALU work hides them.
__global__ void __launch_bounds__(128) attn_fa_kernel(
    const unsigned short* __restrict__ Q,   // [B,H,S,D] bf16, pre-scaled
    const unsigned short* __restrict__ K,   // [B,H,S,D] bf16
    const unsigned short* __restrict__ Vt,  // [B,H,D,S] bf16
    unsigned short* __restrict__ Ao) {      // [B,S,DM] bf16
  __shared__ unsigned short Plds[4][32][32];   // 2 KB per wave
  const int w    = threadIdx.x >> 5;
  const int lane = threadIdx.x & 31;
  const int lh = lane & 15, hf = lane >> 4;
  const int wid = blockIdx.x * 4 + w;
  const int bh = wid >> 6;                 // 64 q-blocks (32 rows) per (b,h)
  const int q0 = (wid & 63) << 5;

  const unsigned short* Qp = Q  + ((size_t)bh * Sn + q0) * Dn;
  const unsigned short* Kp = K  + (size_t)bh * Sn * Dn;
  const unsigned short* Vp = Vt + (size_t)bh * Dn * Sn;

  v16bf qf[2][2];
#pragma unroll
  for (int mt = 0; mt < 2; ++mt)
#pragma unroll
    for (int ks = 0; ks < 2; ++ks)
      qf[mt][ks] = load_fragA(Qp + (size_t)(mt * 16) * Dn + ks * 32, Dn, lane);

  float mrow[2][8], lrow[2][8], alpha[2][8];
#pragma unroll
  for (int mt = 0; mt < 2; ++mt)
#pragma unroll
    for (int v = 0; v < 8; ++v) { mrow[mt][v] = -1e30f; lrow[mt][v] = 0.f; }
  v8f Oacc[2][4] = {};

  const int nkb = (q0 >> 5) + 1;          // causal: only needed key blocks
  for (int kb = 0; kb < nkb; ++kb) {
    const int j0 = kb << 5;
    __builtin_prefetch(Kp + (size_t)(j0 + 32) * Dn, 0, 0);

    // S = Q @ K^T : 2 m-tiles x 2 n-tiles, K-dim = 64 (K-frags reused across m-tiles)
    v8f s[2][2];
#pragma unroll
    for (int t = 0; t < 2; ++t) {
      const v16bf kf0 = load_fragB(Kp + (size_t)(j0 + t * 16) * Dn + 0,  Dn, lane);
      const v16bf kf1 = load_fragB(Kp + (size_t)(j0 + t * 16) * Dn + 32, Dn, lane);
#pragma unroll
      for (int mt = 0; mt < 2; ++mt) {
        v8f st = {};
        st = WMMA_BF16(qf[mt][0], kf0, st);
        st = WMMA_BF16(qf[mt][1], kf1, st);
        s[mt][t] = st;
      }
    }

    // hoist V-fragment loads: independent of S, hidden behind the softmax below
    v16bf vf[4];
#pragma unroll
    for (int dt = 0; dt < 4; ++dt)
      vf[dt] = load_fragB(Vp + (size_t)(dt * 16) * Sn + j0, Sn, lane);

    // causal mask only needed on the diagonal (last) key block; wave-uniform branch
    if (kb == nkb - 1) {
#pragma unroll
      for (int mt = 0; mt < 2; ++mt)
#pragma unroll
        for (int v = 0; v < 8; ++v) {
          const int qr = q0 + mt * 16 + v + (hf << 3);
          s[mt][0][v] = (j0 + lh      > qr) ? -1e30f : s[mt][0][v];
          s[mt][1][v] = (j0 + 16 + lh > qr) ? -1e30f : s[mt][1][v];
        }
    }

    // online softmax; C/D row mapping: row = v + 8*half, col = lane&15
#pragma unroll
    for (int mt = 0; mt < 2; ++mt) {
#pragma unroll
      for (int v = 0; v < 8; ++v) {
        const float s0 = s[mt][0][v], s1 = s[mt][1][v];
        float mc = fmaxf(s0, s1);
        mc = fmaxf(mc, __shfl_xor(mc, 8, 32));
        mc = fmaxf(mc, __shfl_xor(mc, 4, 32));
        mc = fmaxf(mc, __shfl_xor(mc, 2, 32));
        mc = fmaxf(mc, __shfl_xor(mc, 1, 32));
        const float mn = fmaxf(mrow[mt][v], mc);
        const float al = __expf(mrow[mt][v] - mn);
        mrow[mt][v] = mn; alpha[mt][v] = al;
        const float p0 = __expf(s0 - mn);
        const float p1 = __expf(s1 - mn);
        float rs = p0 + p1;
        rs += __shfl_xor(rs, 8, 32);
        rs += __shfl_xor(rs, 4, 32);
        rs += __shfl_xor(rs, 2, 32);
        rs += __shfl_xor(rs, 1, 32);
        lrow[mt][v] = lrow[mt][v] * al + rs;
        // stage P (bf16) through LDS to re-layout D-tile -> A-fragment
        Plds[w][mt * 16 + v + (hf << 3)][lh]      = f2bf(p0);
        Plds[w][mt * 16 + v + (hf << 3)][16 + lh] = f2bf(p1);
      }
    }
    // same-wave LDS ops are in-order; fence the compiler + DS counter
    asm volatile("s_wait_dscnt 0" ::: "memory");
    const v16bf pf0 = load_fragA(&Plds[w][0][0],  32, lane);
    const v16bf pf1 = load_fragA(&Plds[w][16][0], 32, lane);

    // O = diag(alpha)*O + P @ V   (V-frags reused across both m-tiles)
#pragma unroll
    for (int dt = 0; dt < 4; ++dt) {
#pragma unroll
      for (int mt = 0; mt < 2; ++mt) {
        v8f c = Oacc[mt][dt];
#pragma unroll
        for (int v = 0; v < 8; ++v) c[v] *= alpha[mt][v];
        Oacc[mt][dt] = WMMA_BF16(mt == 0 ? pf0 : pf1, vf[dt], c);
      }
    }
  }

  // epilogue: normalize and write bf16 [B,S,DM] for the out-projection
  const int b = bh >> 4, h = bh & 15;
#pragma unroll
  for (int mt = 0; mt < 2; ++mt)
#pragma unroll
    for (int dt = 0; dt < 4; ++dt)
#pragma unroll
      for (int v = 0; v < 8; ++v) {
        const int qr = q0 + mt * 16 + v + (hf << 3);
        Ao[((size_t)(b * Sn + qr)) * DMn + h * Dn + dt * 16 + lh] =
            f2bf(Oacc[mt][dt][v] / lrow[mt][v]);
      }
}

// ---------------- output projection GEMM ----------------
// One wave computes a 16(M) x 64(N) strip, ping-pong pipelined like qkv_gemm.
__global__ void __launch_bounds__(128) out_gemm_kernel(
    const unsigned short* __restrict__ A,   // [B*S][DM] bf16 (attention out)
    const unsigned short* __restrict__ Wt,  // [DM][DM] bf16 (w_out transposed)
    const float* __restrict__ bias,         // [DM]
    float* __restrict__ out) {              // [B*S][DM] f32
  const int lane = threadIdx.x & 31;
  const int wid  = blockIdx.x * 4 + (threadIdx.x >> 5);
  const int mt = wid >> 4, ns = wid & 15;        // 256 x 16 strips
  const int m0 = mt << 4, n0 = ns << 6;
  const int lh = lane & 15, hf = lane >> 4;
  const unsigned short* Ap = A + (size_t)m0 * DMn;

  v8f acc[4] = {};
  v16bf a0 = load_fragA(Ap, DMn, lane);
  v16bf b0[4];
#pragma unroll
  for (int nt = 0; nt < 4; ++nt)
    b0[nt] = load_fragB(Wt + (size_t)(n0 + nt * 16) * DMn, DMn, lane);

  for (int kk = 0; kk < DMn; kk += 64) {
    const v16bf a1 = load_fragA(Ap + kk + 32, DMn, lane);
    v16bf b1[4];
#pragma unroll
    for (int nt = 0; nt < 4; ++nt)
      b1[nt] = load_fragB(Wt + (size_t)(n0 + nt * 16) * DMn + kk + 32, DMn, lane);
#pragma unroll
    for (int nt = 0; nt < 4; ++nt)
      acc[nt] = WMMA_BF16(a0, b0[nt], acc[nt]);
    if (kk + 64 < DMn) {
      a0 = load_fragA(Ap + kk + 64, DMn, lane);
#pragma unroll
      for (int nt = 0; nt < 4; ++nt)
        b0[nt] = load_fragB(Wt + (size_t)(n0 + nt * 16) * DMn + kk + 64, DMn, lane);
    }
#pragma unroll
    for (int nt = 0; nt < 4; ++nt)
      acc[nt] = WMMA_BF16(a1, b1[nt], acc[nt]);
  }

#pragma unroll
  for (int nt = 0; nt < 4; ++nt) {
    const float bv = bias[n0 + nt * 16 + lh];
#pragma unroll
    for (int v = 0; v < 8; ++v)
      out[(size_t)(m0 + v + (hf << 3)) * DMn + n0 + nt * 16 + lh] = acc[nt][v] + bv;
  }
}

// ---------------- launcher ----------------
extern "C" void kernel_launch(void* const* d_in, const int* in_sizes, int n_in,
                              void* d_out, int out_size, void* d_ws, size_t ws_size,
                              hipStream_t stream) {
  const float* x    = (const float*)d_in[0];
  const float* wqkv = (const float*)d_in[1];
  const float* bqkv = (const float*)d_in[2];
  const float* wout = (const float*)d_in[3];
  const float* bout = (const float*)d_in[4];
  float* out = (float*)d_out;

  // workspace layout (48 MB total)
  char* ws = (char*)d_ws;
  unsigned short* xb    = (unsigned short*)(ws);                        //  8 MB  x bf16
  unsigned short* wqkvT = (unsigned short*)(ws + (size_t) 8 * 1048576); //  6 MB
  unsigned short* woutT = (unsigned short*)(ws + (size_t)14 * 1048576); //  2 MB
  unsigned short* Qb    = (unsigned short*)(ws + (size_t)16 * 1048576); //  8 MB
  unsigned short* Kb    = (unsigned short*)(ws + (size_t)24 * 1048576); //  8 MB
  unsigned short* Vtb   = (unsigned short*)(ws + (size_t)32 * 1048576); //  8 MB
  unsigned short* Ab    = (unsigned short*)(ws + (size_t)40 * 1048576); //  8 MB

  // 1) precision conversion (+ weight transposes for contiguous B-fragments)
  cvt_bf16_kernel<<<16384, 256, 0, stream>>>(x, xb, Bn * Sn * DMn);
  cvt_T_bf16_kernel<<<12288, 256, 0, stream>>>(wqkv, wqkvT, DMn, 3 * DMn);
  cvt_T_bf16_kernel<<<4096, 256, 0, stream>>>(wout, woutT, DMn, DMn);

  // 2) QKV projection: 256*48 wave-strips, 4 waves/block
  qkv_gemm_kernel<<<3072, 128, 0, stream>>>(xb, wqkvT, bqkv, Qb, Kb, Vtb);

  // 3) flash attention: B*H*(S/32) = 2048 waves, 4 waves/block
  attn_fa_kernel<<<512, 128, 0, stream>>>(Qb, Kb, Vtb, Ab);

  // 4) output projection: 256*16 wave-strips
  out_gemm_kernel<<<1024, 128, 0, stream>>>(Ab, woutT, bout, out);
}
